// Diffusion_80891414053479
// MI455X (gfx1250) — compile-verified, matrix operand order checked by reading, not run on hardware
//
#include <hip/hip_runtime.h>
#include <hip/hip_bf16.h>
#include <stdint.h>

// ---------------------------------------------------------------------------
// Spectral diffusion: out = evecs @ diag(exp(-t*evals)) @ evecs^T @ x
// N = 8192, C = 256.  Split-bf16 (hi+lo) WMMA GEMM, fp32 accumulate.
// B operands stored in WMMA-fragment-major layout (2x global b128 per frag).
// Double-buffered LDS for the A (evecs) tile, split fp32->bf16 on the fly.
// Steady-state loop is unconditional (last chunk peeled) to avoid acc PHIs.
// ---------------------------------------------------------------------------

#define NMAT 8192
#define CCH  256
#define MT   32      // workgroup M tile  -> 256 WGs per GEMM
#define KC   32      // K chunk (== WMMA K for bf16)
#define ASTR 40      // LDS row stride in bf16 elems (80B: 16B-aligned, conflict-free)

typedef __attribute__((ext_vector_type(16))) __bf16 v16bf;
typedef __attribute__((ext_vector_type(8)))  float  v8f;

union ABFrag { v16bf v; unsigned u[8]; };

__device__ __forceinline__ unsigned short f2bf_rne(float f) {
    unsigned u = __builtin_bit_cast(unsigned, f);
    unsigned r = u + 0x7FFFu + ((u >> 16) & 1u);   // round-to-nearest-even
    return (unsigned short)(r >> 16);
}

__device__ __forceinline__ void split_bf16(float f, unsigned short& h, unsigned short& l) {
    unsigned short hb = f2bf_rne(f);
    float fh = __builtin_bit_cast(float, (unsigned)hb << 16);
    h = hb;
    l = f2bf_rne(f - fh);
}

// fragment-major index for B element pair (rows 2t,2t+1, col n)
__device__ __forceinline__ size_t bfrag_index(int t, int n) {
    int kchunk = t >> 4;                 // 16 pairs per K-chunk of 32
    int tl     = t & 15;
    int hb     = (tl >> 2) & 1;
    int j      = (tl & 3) + ((tl >= 8) ? 4 : 0);
    int laneb  = (n & 15) + hb * 16;
    return ((((size_t)kchunk * 16 + (n >> 4)) * 32 + laneb) << 3) + j;
}

// ---------------------------------------------------------------------------
// prep: coefs[k] = exp(-evals[k]*clamp(t,1e-8));
//       x -> fragment-major packed bf16 hi/lo (one thread per packed dword)
// ---------------------------------------------------------------------------
__global__ __launch_bounds__(256)
void prep_kernel(const float* __restrict__ x, const float* __restrict__ evals,
                 const float* __restrict__ tparam,
                 float* __restrict__ coefs,
                 unsigned* __restrict__ xhi, unsigned* __restrict__ xlo) {
    int i = blockIdx.x * 256 + threadIdx.x;        // 0 .. (N/2)*C - 1
    int j     = i & 7;
    int laneb = (i >> 3) & 31;
    int ntile = (i >> 8) & 15;
    int kc    = i >> 12;
    int hb    = laneb >> 4;
    int n     = ntile * 16 + (laneb & 15);
    int tl    = j + ((j >= 4) ? 4 : 0) + hb * 4;
    int t     = kc * 16 + tl;                       // row pair index
    float f0 = x[(size_t)(2 * t) * CCH + n];
    float f1 = x[(size_t)(2 * t) * CCH + CCH + n];
    unsigned short h0, l0, h1, l1;
    split_bf16(f0, h0, l0);
    split_bf16(f1, h1, l1);
    xhi[i] = (unsigned)h0 | ((unsigned)h1 << 16);
    xlo[i] = (unsigned)l0 | ((unsigned)l1 << 16);
    if (i < NMAT) {
        float tt = tparam[0];
        tt = (tt < 1e-8f) ? 1e-8f : tt;
        coefs[i] = expf(-evals[i] * tt);
    }
}

// ---------------------------------------------------------------------------
// Split-bf16 GEMM:  D[m,n] = sum_k A[m,k]*B[k,n]
//   TRANS_A=1 : A[m,k] = evecs[k][m]   (pass 1: evecs^T @ x)
//   TRANS_A=0 : A[m,k] = evecs[m][k]   (pass 2: evecs @ xs)
//   SCALE_SPLIT=1: epilogue scales rows by coefs[], writes fragment-major bf16.
//   SCALE_SPLIT=0: epilogue writes fp32.
// Block = 256 threads = 8 waves; WG tile 32x256; wave tile 16x64.
// ---------------------------------------------------------------------------
template <bool TRANS_A, bool SCALE_SPLIT>
__global__ __launch_bounds__(256)
void gemm_split_bf16(const float* __restrict__ evecs,
                     const unsigned* __restrict__ Bhi,
                     const unsigned* __restrict__ Blo,
                     const float* __restrict__ coefs,
                     unsigned* __restrict__ OutHi,
                     unsigned* __restrict__ OutLo,
                     float* __restrict__ OutF) {
    __shared__ __align__(16) unsigned short Ah[2][MT * ASTR];
    __shared__ __align__(16) unsigned short Al[2][MT * ASTR];

    const int tid  = threadIdx.x;
    const int wave = tid >> 5;
    const int lane = tid & 31;
    const int half = lane >> 4;          // 0: lanes 0-15, 1: lanes 16-31
    const int ln   = lane & 15;
    const int wm   = (wave >> 2) * 16;   // wave row offset in WG tile
    const int wn   = (wave & 3) * 64;    // wave col offset
    const int m0   = blockIdx.x * MT;

    v8f acc[4] = {};
    float stage[4];

    // ---- staged global load of one 32x32 A tile (1 float4 per thread)
    auto load_stage = [&](int kt) {
        const float* src;
        if (TRANS_A) {
            int kk = tid >> 3, mg = tid & 7;
            src = evecs + (size_t)(kt + kk) * NMAT + m0 + mg * 4;   // A[m][kk]=evecs[kk][m]
        } else {
            int m = tid >> 3, kg = tid & 7;
            src = evecs + (size_t)(m0 + m) * NMAT + kt + kg * 4;
        }
        float4 f = *(const float4*)src;
        stage[0] = f.x; stage[1] = f.y; stage[2] = f.z; stage[3] = f.w;
        if (kt + 2 * KC < NMAT)
            __builtin_prefetch(src + (TRANS_A ? (size_t)KC * NMAT : (size_t)KC), 0, 1);
    };
    // ---- split staged values and store to LDS buffer
    auto store_stage = [&](int buf) {
        #pragma unroll
        for (int e = 0; e < 4; ++e) {
            unsigned short h, l;
            split_bf16(stage[e], h, l);
            int idx;
            if (TRANS_A) { int kk = tid >> 3, mg = tid & 7; idx = (mg * 4 + e) * ASTR + kk; }
            else         { int m  = tid >> 3, kg = tid & 7; idx = m * ASTR + kg * 4 + e; }
            Ah[buf][idx] = h;
            Al[buf][idx] = l;
        }
    };
    // ---- one K-chunk of WMMA work from LDS buffer `buf`, global B at kt
    auto compute = [&](int buf, int kt) {
        // A fragments (16x32 bf16, ISA layout): two b128 LDS loads each
        ABFrag ah, al;
        {
            int m = wm + ln;
            const unsigned short* pa = &Ah[buf][m * ASTR + half * 8];
            const unsigned short* pl = &Al[buf][m * ASTR + half * 8];
            uint4 q0 = *(const uint4*)pa;
            uint4 q1 = *(const uint4*)(pa + 16);
            ah.u[0] = q0.x; ah.u[1] = q0.y; ah.u[2] = q0.z; ah.u[3] = q0.w;
            ah.u[4] = q1.x; ah.u[5] = q1.y; ah.u[6] = q1.z; ah.u[7] = q1.w;
            uint4 r0 = *(const uint4*)pl;
            uint4 r1 = *(const uint4*)(pl + 16);
            al.u[0] = r0.x; al.u[1] = r0.y; al.u[2] = r0.z; al.u[3] = r0.w;
            al.u[4] = r1.x; al.u[5] = r1.y; al.u[6] = r1.z; al.u[7] = r1.w;
        }
        // all B fragments up front: fragment-major, 2x global b128 each
        ABFrag bh[4], bl[4];
        const size_t chunk_base = ((size_t)(kt >> 5) * 16 + (wn >> 4)) * 32;
        #pragma unroll
        for (int fc = 0; fc < 4; ++fc) {
            size_t fb = ((chunk_base + (size_t)fc * 32 + lane) << 3);
            uint4 q0 = *(const uint4*)&Bhi[fb];
            uint4 q1 = *(const uint4*)&Bhi[fb + 4];
            bh[fc].u[0] = q0.x; bh[fc].u[1] = q0.y; bh[fc].u[2] = q0.z; bh[fc].u[3] = q0.w;
            bh[fc].u[4] = q1.x; bh[fc].u[5] = q1.y; bh[fc].u[6] = q1.z; bh[fc].u[7] = q1.w;
            uint4 r0 = *(const uint4*)&Blo[fb];
            uint4 r1 = *(const uint4*)&Blo[fb + 4];
            bl[fc].u[0] = r0.x; bl[fc].u[1] = r0.y; bl[fc].u[2] = r0.z; bl[fc].u[3] = r0.w;
            bl[fc].u[4] = r1.x; bl[fc].u[5] = r1.y; bl[fc].u[6] = r1.z; bl[fc].u[7] = r1.w;
        }
        // 12 back-to-back WMMAs (3-term split product)
        #pragma unroll
        for (int fc = 0; fc < 4; ++fc)
            acc[fc] = __builtin_amdgcn_wmma_f32_16x16x32_bf16(
                false, ah.v, false, bh[fc].v, (short)0, acc[fc], false, false);
        #pragma unroll
        for (int fc = 0; fc < 4; ++fc)
            acc[fc] = __builtin_amdgcn_wmma_f32_16x16x32_bf16(
                false, ah.v, false, bl[fc].v, (short)0, acc[fc], false, false);
        #pragma unroll
        for (int fc = 0; fc < 4; ++fc)
            acc[fc] = __builtin_amdgcn_wmma_f32_16x16x32_bf16(
                false, al.v, false, bh[fc].v, (short)0, acc[fc], false, false);
    };

    load_stage(0);
    store_stage(0);
    int cur = 0;

    // steady state: unconditional body (no acc PHIs), last chunk peeled
    for (int kt = 0; kt < NMAT - KC; kt += KC) {
        load_stage(kt + KC);     // next A tile: global loads issued, no wait yet
        __syncthreads();         // buffer `cur` ready for everyone
        compute(cur, kt);        // 12 WMMAs issued before the stage wait lands
        store_stage(cur ^ 1);    // convert + ds_store behind the WMMAs
        cur ^= 1;
    }
    __syncthreads();
    compute(cur, NMAT - KC);

    // ---- epilogue. D layout: VGPR r -> M = r + half*8, lane col = ln.
    const int mbase = m0 + wm + half * 8;
    #pragma unroll
    for (int fc = 0; fc < 4; ++fc) {
        int n = wn + fc * 16 + ln;
        v8f cacc = acc[fc];
        if (SCALE_SPLIT) {
            #pragma unroll
            for (int r = 0; r < 8; r += 2) {
                int m = mbase + r;
                float c0 = coefs[m] * cacc[r];
                float c1 = coefs[m + 1] * cacc[r + 1];
                unsigned short h0, l0, h1, l1;
                split_bf16(c0, h0, l0);
                split_bf16(c1, h1, l1);
                size_t oi = bfrag_index(m >> 1, n);   // fragment-major for pass 2
                OutHi[oi] = (unsigned)h0 | ((unsigned)h1 << 16);
                OutLo[oi] = (unsigned)l0 | ((unsigned)l1 << 16);
            }
        } else {
            #pragma unroll
            for (int r = 0; r < 8; ++r)
                OutF[(size_t)(mbase + r) * CCH + n] = cacc[r];
        }
    }
}

// ---------------------------------------------------------------------------
extern "C" void kernel_launch(void* const* d_in, const int* in_sizes, int n_in,
                              void* d_out, int out_size, void* d_ws, size_t ws_size,
                              hipStream_t stream) {
    const float* x      = (const float*)d_in[0];
    const float* evals  = (const float*)d_in[1];
    const float* evecs  = (const float*)d_in[2];
    const float* tparam = (const float*)d_in[3];
    float* out = (float*)d_out;

    // workspace layout
    char* ws = (char*)d_ws;
    const size_t pk_elems = (size_t)(NMAT / 2) * CCH;   // packed dwords per array
    float*    coefs = (float*)ws;                        // 32 KB
    unsigned* x_hi  = (unsigned*)(ws + (64 << 10));      // 4 MB each
    unsigned* x_lo  = x_hi + pk_elems;
    unsigned* xs_hi = x_lo + pk_elems;
    unsigned* xs_lo = xs_hi + pk_elems;

    // pass 0: split x into fragment-major bf16 hi/lo, compute spectral coefs
    prep_kernel<<<(NMAT / 2) * CCH / 256, 256, 0, stream>>>(x, evals, tparam,
                                                            coefs, x_hi, x_lo);
    // pass 1: xs = diag(coefs) * (evecs^T @ x), fragment-major bf16 hi/lo out
    gemm_split_bf16<true, true><<<NMAT / MT, 256, 0, stream>>>(
        evecs, x_hi, x_lo, coefs, xs_hi, xs_lo, nullptr);
    // pass 2: out = evecs @ xs, fp32 output
    gemm_split_bf16<false, false><<<NMAT / MT, 256, 0, stream>>>(
        evecs, xs_hi, xs_lo, nullptr, nullptr, nullptr, out);
}
